// CorrelationModule_27049704030772
// MI455X (gfx1250) — compile-verified, matrix operand order checked by reading, not run on hardware
//
#include <hip/hip_runtime.h>

typedef __attribute__((ext_vector_type(16))) _Float16 v16h;
typedef __attribute__((ext_vector_type(8)))  _Float16 v8h;
typedef __attribute__((ext_vector_type(8)))  float    v8f;

// Build a WMMA 16-bit A/B fragment for the 16x16x32 f16 op.
// Per CDNA5 ISA: lane l -> row/col (l&15); kh=(l>>4)*8; elems 0..7 = K kh+0..7,
// elems 8..15 = K kh+16..23.  Caller passes the two 8-half contiguous segments.
static __device__ inline v16h frag_ld(const _Float16* lo, const _Float16* hi) {
  v8h a = *(const v8h*)lo;
  v8h b = *(const v8h*)hi;
  v16h r;
#pragma unroll
  for (int i = 0; i < 8; ++i) { r[i] = a[i]; r[i + 8] = b[i]; }
  return r;
}

static __device__ inline v16h frag_zero() {
  v16h r;
#pragma unroll
  for (int i = 0; i < 16; ++i) r[i] = (_Float16)0;
  return r;
}

// ---------------------------------------------------------------------------
// Kernel 1: BN + ReLU + NCHW->NHWC transpose, LDS-staged for coalescing.
// One block per (feat, b, h): reads 128c x 128w f32 coalesced along w,
// writes act[feat][b][h][w][c] f16 coalesced along c.
// ---------------------------------------------------------------------------
__launch_bounds__(256)
__global__ void bnrelu_kernel(const float* __restrict__ x,
                              const float* __restrict__ gamma,
                              const float* __restrict__ beta,
                              const float* __restrict__ mean,
                              const float* __restrict__ var,
                              _Float16* __restrict__ act) {
  __shared__ __align__(16) _Float16 tile[128 * 128];  // [w][c], 32 KB

  int bx = blockIdx.x;                 // (f*4 + b)*64 + h
  int h = bx & 63;
  int b = (bx >> 6) & 3;
  int f = bx >> 8;
  int tid = threadIdx.x;

  // Read phase: thread -> channel row c = tid>>1, w-segment of 64 floats.
  int c = tid >> 1;
  int wseg = (tid & 1) * 64;
  float inv = rsqrtf(var[c] + 1e-5f);
  float s = gamma[c] * inv;
  float mu = mean[c];
  float be = beta[c];
  const float* src =
      x + (((size_t)b * 512 + f * 128 + c) * 64 + h) * 128 + wseg;
#pragma unroll
  for (int i = 0; i < 64; i += 4) {
    float4 v = *(const float4*)(src + i);
    tile[(wseg + i + 0) * 128 + c] = (_Float16)fmaxf((v.x - mu) * s + be, 0.f);
    tile[(wseg + i + 1) * 128 + c] = (_Float16)fmaxf((v.y - mu) * s + be, 0.f);
    tile[(wseg + i + 2) * 128 + c] = (_Float16)fmaxf((v.z - mu) * s + be, 0.f);
    tile[(wseg + i + 3) * 128 + c] = (_Float16)fmaxf((v.w - mu) * s + be, 0.f);
  }
  __syncthreads();

  // Write phase: thread -> pixel w = tid>>1, c-segment of 64 halfs (128 B).
  int w = tid >> 1;
  int cseg = (tid & 1) * 64;
  const uint4* ts = (const uint4*)&tile[w * 128 + cseg];
  uint4* dst = (uint4*)(act + ((size_t)bx * 128 + w) * 128 + cseg);
#pragma unroll
  for (int i = 0; i < 8; ++i) dst[i] = ts[i];
}

// ---------------------------------------------------------------------------
// Kernel 2: weight transform OIHW f32 -> [tap][co][ci] f16
// ---------------------------------------------------------------------------
__global__ void wprep_kernel(const float* __restrict__ wsrc,
                             _Float16* __restrict__ wtap) {
  unsigned idx = blockIdx.x * blockDim.x + threadIdx.x;  // 9*128*128
  int ci = idx & 127;
  int co = (idx >> 7) & 127;
  int tap = idx >> 14;
  wtap[idx] = (_Float16)wsrc[((size_t)co * 128 + ci) * 9 + tap];
}

// ---------------------------------------------------------------------------
// Kernel 3: 3x3 conv as implicit GEMM (M=128 px row, N=128 cout, K=9*128),
// software-pipelined (register prefetch of next chunk over WMMA compute),
// fused channel-L2 normalization epilogue.  One block per (feat,b,h).
// ---------------------------------------------------------------------------
__launch_bounds__(256)
__global__ void convnorm_kernel(const _Float16* __restrict__ act,
                                const _Float16* __restrict__ wtap,
                                _Float16* __restrict__ normf) {
  __shared__ __align__(16) _Float16 ldsA[128 * 32];  // [w][ci32]
  __shared__ __align__(16) _Float16 ldsB[128 * 32];  // [co][ci32]

  int bx = blockIdx.x;
  int h = bx & 63;
  int b = (bx >> 6) & 3;
  int f = bx >> 8;
  int tid = threadIdx.x;
  int lane = tid & 31;
  int wave = tid >> 5;

  const _Float16* actbase = act + (size_t)(f * 4 + b) * 64 * 128 * 128;

  v8f acc[8];
#pragma unroll
  for (int nf = 0; nf < 8; ++nf)
#pragma unroll
    for (int v = 0; v < 8; ++v) acc[nf][v] = 0.0f;

  int arow = tid >> 1;        // 0..127 : pixel w (A tile) / cout (B tile)
  int aseg = (tid & 1) * 16;  // half-row segment in halfs

  int r = lane & 15;
  int kh = (lane >> 4) * 8;
  int m0 = wave * 16;

  // chunk index c in [0,36): tap = c>>2, k0 = (c&3)*32
  auto load_chunk = [&](int c, uint4& a0, uint4& a1, uint4& b0, uint4& b1) {
    int tap = c >> 2;
    int k0 = (c & 3) << 5;
    int ky = tap / 3, kx = tap - ky * 3;
    int hin = h + ky - 1;
    int win = arow + kx - 1;
    bool avalid = (hin >= 0) && (hin < 64) && (win >= 0) && (win < 128);
    uint4 z = {0u, 0u, 0u, 0u};
    a0 = z;
    a1 = z;
    if (avalid) {
      const uint4* p = (const uint4*)(actbase +
                                      ((size_t)hin * 128 + win) * 128 + k0 +
                                      aseg);
      a0 = p[0];
      a1 = p[1];
    }
    const uint4* pB =
        (const uint4*)(wtap + (size_t)(tap * 128 + arow) * 128 + k0 + aseg);
    b0 = pB[0];
    b1 = pB[1];
  };

  uint4* dA = (uint4*)&ldsA[arow * 32 + aseg];
  uint4* dB = (uint4*)&ldsB[arow * 32 + aseg];

  // Prologue: chunk 0 into LDS.
  uint4 ca0, ca1, cb0, cb1;
  load_chunk(0, ca0, ca1, cb0, cb1);
  dA[0] = ca0;
  dA[1] = ca1;
  dB[0] = cb0;
  dB[1] = cb1;
  __syncthreads();

#pragma unroll 1
  for (int c = 0; c < 36; ++c) {
    // Prefetch next chunk into registers (overlaps with WMMA below).
    uint4 na0, na1, nb0, nb1;
    if (c + 1 < 36) load_chunk(c + 1, na0, na1, nb0, nb1);

    // Compute current chunk from LDS.
    v16h afrag = frag_ld(&ldsA[(m0 + r) * 32 + kh],
                         &ldsA[(m0 + r) * 32 + kh + 16]);
#pragma unroll
    for (int nf = 0; nf < 8; ++nf) {
      int col = nf * 16 + r;
      v16h bfrag = frag_ld(&ldsB[col * 32 + kh], &ldsB[col * 32 + kh + 16]);
      acc[nf] = __builtin_amdgcn_wmma_f32_16x16x32_f16(
          false, afrag, false, bfrag, (short)0, acc[nf], false, false);
    }
    __syncthreads();
    if (c + 1 < 36) {
      dA[0] = na0;
      dA[1] = na1;
      dB[0] = nb0;
      dB[1] = nb1;
    }
    __syncthreads();
  }

  // --- fused L2 normalization over the 128 couts of each pixel row ---
  // C layout: lane l holds col nf*16+(l&15), rows v + 8*(l>>4).
  float ss[8];
#pragma unroll
  for (int v = 0; v < 8; ++v) {
    float s = 0.0f;
#pragma unroll
    for (int nf = 0; nf < 8; ++nf) s += acc[nf][v] * acc[nf][v];
    ss[v] = s;
  }
#pragma unroll
  for (int off = 1; off < 16; off <<= 1) {
#pragma unroll
    for (int v = 0; v < 8; ++v) ss[v] += __shfl_xor(ss[v], off, 32);
  }

  _Float16* outb = normf + (size_t)((f * 4 + b) * 64 + h) * 128 * 128;
  int hhalf = lane >> 4;
#pragma unroll
  for (int v = 0; v < 8; ++v) {
    float rn = rsqrtf(ss[v]);
    int wpix = m0 + v + 8 * hhalf;
#pragma unroll
    for (int nf = 0; nf < 8; ++nf) {
      int co = nf * 16 + r;
      outb[(size_t)wpix * 128 + co] = (_Float16)(acc[nf][v] * rn);
    }
  }
}

// ---------------------------------------------------------------------------
// Kernel 4: dilated 9x9 correlation as banded WMMA GEMM.
// Block = (pair, b, h, py); wave ww = w-tile [16*ww, 16*ww+16).
// D = f1_tile(16x128) x f2_winT(128x32); out[r][px] = D[r][r + 2*px].
// ---------------------------------------------------------------------------
__launch_bounds__(256)
__global__ void corr_kernel(const _Float16* __restrict__ normf,
                            float* __restrict__ out) {
  __shared__ __align__(16) float dscr[8][16 * 32];

  int bx = blockIdx.x;
  int py = bx % 9;
  int t = bx / 9;
  int h = t & 63;
  t >>= 6;
  int b = t & 3;
  int pair = t >> 2;

  int tid = threadIdx.x;
  int lane = tid & 31;
  int wave = tid >> 5;
  int w0 = wave * 16;
  int r = lane & 15;
  int kh = (lane >> 4) * 8;

  const _Float16* f1 = normf + (size_t)((pair * 4 + b) * 64 + h) * 128 * 128;
  int h2 = h + 2 * py - 8;
  bool h2v = (h2 >= 0) && (h2 < 64);
  const _Float16* f2row =
      normf + (size_t)(((pair + 1) * 4 + b) * 64 + (h2v ? h2 : 0)) * 128 * 128;

  // A fragments: pixel rows, K = ci, 4 chunks of 32 (contiguous in NHWC).
  v16h a[4];
#pragma unroll
  for (int kc = 0; kc < 4; ++kc) {
    const _Float16* p = f1 + (size_t)(w0 + r) * 128 + kc * 32 + kh;
    a[kc] = frag_ld(p, p + 16);
  }

  v8f c0[2];
#pragma unroll
  for (int v = 0; v < 8; ++v) { c0[0][v] = 0.0f; c0[1][v] = 0.0f; }

#pragma unroll
  for (int nt = 0; nt < 2; ++nt) {
    int j = nt * 16 + r;  // column of the 32-wide f2 window
    int wp = w0 - 8 + j;  // absolute f2 pixel
    bool valid = h2v && (wp >= 0) && (wp < 128);
    const _Float16* pcol = f2row + (size_t)(valid ? wp : 0) * 128;
#pragma unroll
    for (int kc = 0; kc < 4; ++kc) {
      v16h bf = frag_zero();
      if (valid) {
        const _Float16* p = pcol + kc * 32 + kh;
        bf = frag_ld(p, p + 16);
      }
      c0[nt] = __builtin_amdgcn_wmma_f32_16x16x32_f16(
          false, a[kc], false, bf, (short)0, c0[nt], false, false);
    }
  }

  // Spill D (16x32) to LDS, then gather the dilated band.
  int hhalf = lane >> 4;
#pragma unroll
  for (int nt = 0; nt < 2; ++nt)
#pragma unroll
    for (int v = 0; v < 8; ++v) {
      int m = v + 8 * hhalf;
      dscr[wave][m * 32 + nt * 16 + r] = c0[nt][v];
    }
  __syncthreads();

  // Band extraction, w-fastest for coalesced 4B stores (16 consecutive w per
  // half-wave -> two 64B segments per store wavefront).
  for (int idx = lane; idx < 16 * 9; idx += 32) {
    int px = idx >> 4;   // 0..8
    int rr = idx & 15;   // 0..15, consecutive lanes -> consecutive w
    float val = dscr[wave][rr * 32 + rr + 2 * px];
    out[((size_t)(b * 243 + pair * 81 + py * 9 + px) * 64 + h) * 128 + w0 +
        rr] = val;
  }
}

// ---------------------------------------------------------------------------
extern "C" void kernel_launch(void* const* d_in, const int* in_sizes, int n_in,
                              void* d_out, int out_size, void* d_ws,
                              size_t ws_size, hipStream_t stream) {
  const float* x = (const float*)d_in[0];
  const float* gamma = (const float*)d_in[1];
  const float* beta = (const float*)d_in[2];
  const float* mean = (const float*)d_in[3];
  const float* var = (const float*)d_in[4];
  const float* wsrc = (const float*)d_in[5];
  float* out = (float*)d_out;

  char* ws = (char*)d_ws;
  _Float16* act = (_Float16*)(ws);                               // 32 MB
  _Float16* normf = (_Float16*)(ws + (size_t)32 * 1024 * 1024);  // 32 MB
  _Float16* wtap = (_Float16*)(ws + (size_t)64 * 1024 * 1024);   // 288 KB

  // one block per (feat, b, h)
  bnrelu_kernel<<<4 * 4 * 64, 256, 0, stream>>>(x, gamma, beta, mean, var,
                                                act);
  // 9*128*128 = 147456 elements
  wprep_kernel<<<576, 256, 0, stream>>>(wsrc, wtap);
  // one block per (feat, b, h)
  convnorm_kernel<<<4 * 4 * 64, 256, 0, stream>>>(act, wtap, normf);
  // one block per (pair, b, h, py)
  corr_kernel<<<3 * 4 * 64 * 9, 256, 0, stream>>>(normf, out);
}
